// Lstm_28518582846299
// MI455X (gfx1250) — matrix operand order repaired and checked by reading, not compiled, not run on hardware
//
#include <hip/hip_runtime.h>
#include <hip/hip_bf16.h>
#include <math.h>

// LSTM: SEQ=8192, E=256, H=512, batch=1.
// Phase 1 (parallel): Gx = Ebf @ Wih^T + b_ih + b_hh  via bf16 WMMA GEMM
//                     (double-buffered K loop to hide global load latency).
// Phase 2 (sequential): 8 persistent WGs, W_hh slices DMA'd into LDS by the
// Tensor Data Mover (tensor_load_to_lds, TENSORcnt) with row padding chosen
// for conflict-free A-fragment ds_load_b128; per-step matvec via
// v_wmma_f32_16x16x32_bf16 with h broadcast into all 16 B columns;
// cross-WG sync via fence+counter; global_prefetch_b8 for next-step Gx.

#define SEQ_LEN 8192
#define EMB_DIM 256
#define HID_DIM 512
#define GATES   (4 * HID_DIM)   // 2048

#define RWG        8            // workgroups in recurrent phase
#define JPER       64           // h elements owned per WG (512/8)
#define ROW_BYTES  1040         // 512 bf16 (1024B) + 16B TDM pad -> bank-conflict free
#define QBLK_BYTES (64 * ROW_BYTES)  // one 64-row quadrant block in LDS (66560)

typedef __attribute__((ext_vector_type(16))) __bf16 v16bf;
typedef __attribute__((ext_vector_type(8)))  __bf16 v8bf;
typedef __attribute__((ext_vector_type(8)))  float  v8f;
typedef __attribute__((ext_vector_type(4)))  unsigned int v4u;
typedef __attribute__((ext_vector_type(8)))  int v8i;
typedef __attribute__((ext_vector_type(4)))  int v4i;

__device__ __forceinline__ v16bf make_v16(v8bf lo, v8bf hi) {
    v16bf r;
#pragma unroll
    for (int i = 0; i < 8; i++) { r[i] = lo[i]; r[8 + i] = hi[i]; }
    return r;
}

__device__ __forceinline__ float sigmf(float x) {
    return 1.0f / (1.0f + __expf(-x));
}

// ---------------- fp32 -> bf16 conversion ----------------
__global__ void cvt_bf16(const float* __restrict__ in, __bf16* __restrict__ out, int n) {
    int i = blockIdx.x * blockDim.x + threadIdx.x;
    if (i < n) out[i] = (__bf16)in[i];
}

// ---------------- embedding gather -> bf16 ----------------
__global__ void embed_gather(const int* __restrict__ x, const float* __restrict__ emb,
                             __bf16* __restrict__ Ebf) {
    int t = blockIdx.x;
    int row = x[t];
    const float4* src = (const float4*)(emb + (size_t)row * EMB_DIM);
    float4 v = src[threadIdx.x];            // 64 threads * 4 floats = 256
    __bf16* dst = Ebf + (size_t)t * EMB_DIM + threadIdx.x * 4;
    dst[0] = (__bf16)v.x; dst[1] = (__bf16)v.y;
    dst[2] = (__bf16)v.z; dst[3] = (__bf16)v.w;
}

// ---------------- h0 -> bf16 double buffer slot 0 ----------------
__global__ void h_init(const float* __restrict__ h0, __bf16* __restrict__ hbuf) {
    hbuf[threadIdx.x] = (__bf16)h0[threadIdx.x];  // 512 threads
}

// ---------------- Phase 1: pre-gate GEMM ----------------
// [8192 x 256] x [256 x 2048]; block = 8 waves, BM=128, BN=64.
// Wave computes 16(M) x 64(N) with 4 f32 accumulators; K loop step 32,
// software-pipelined (load k+32 fragments while k's WMMAs issue).
__global__ void __launch_bounds__(256) gemm_pregates(
    const __bf16* __restrict__ Ebf, const __bf16* __restrict__ Wih,
    const float* __restrict__ b_ih, const float* __restrict__ b_hh,
    float* __restrict__ Gx)
{
    const int wave = threadIdx.x >> 5;
    const int lane = threadIdx.x & 31;
    const int lo = lane & 15, hi = lane >> 4;
    const int m0 = blockIdx.x * 128 + wave * 16;
    const int n0 = blockIdx.y * 64;

    v8f acc[4];
#pragma unroll
    for (int j = 0; j < 4; j++)
#pragma unroll
        for (int r = 0; r < 8; r++) acc[j][r] = 0.0f;

    // A fragment: lane<16 holds K {0..7,16..23}, lane>=16 {8..15,24..31}
    const __bf16* Arow = Ebf + (size_t)(m0 + lo) * EMB_DIM + hi * 8;
    // B = Wih^T: column n of B == row n of Wih; lane<16 K {0..7,16..23}, lane>=16 {8..15,24..31}
    const __bf16* Brow0 = Wih + (size_t)(n0 +  0 + lo) * EMB_DIM + hi * 8;
    const __bf16* Brow1 = Wih + (size_t)(n0 + 16 + lo) * EMB_DIM + hi * 8;
    const __bf16* Brow2 = Wih + (size_t)(n0 + 32 + lo) * EMB_DIM + hi * 8;
    const __bf16* Brow3 = Wih + (size_t)(n0 + 48 + lo) * EMB_DIM + hi * 8;

#define LD_A(kk)  make_v16(*(const v8bf*)(Arow + (kk)), *(const v8bf*)(Arow + (kk) + 16))
#define LD_B(p,kk) make_v16(*(const v8bf*)((p) + (kk)), *(const v8bf*)((p) + (kk) + 16))

    v16bf A  = LD_A(0);
    v16bf B0 = LD_B(Brow0, 0), B1 = LD_B(Brow1, 0);
    v16bf B2 = LD_B(Brow2, 0), B3 = LD_B(Brow3, 0);

#pragma unroll
    for (int kk = 0; kk < EMB_DIM; kk += 32) {
        v16bf An = A, Bn0 = B0, Bn1 = B1, Bn2 = B2, Bn3 = B3;
        if (kk + 32 < EMB_DIM) {            // folds at compile time (unrolled)
            An  = LD_A(kk + 32);
            Bn0 = LD_B(Brow0, kk + 32); Bn1 = LD_B(Brow1, kk + 32);
            Bn2 = LD_B(Brow2, kk + 32); Bn3 = LD_B(Brow3, kk + 32);
        }
        acc[0] = __builtin_amdgcn_wmma_f32_16x16x32_bf16(false, A, false, B0, (short)0, acc[0], false, false);
        acc[1] = __builtin_amdgcn_wmma_f32_16x16x32_bf16(false, A, false, B1, (short)0, acc[1], false, false);
        acc[2] = __builtin_amdgcn_wmma_f32_16x16x32_bf16(false, A, false, B2, (short)0, acc[2], false, false);
        acc[3] = __builtin_amdgcn_wmma_f32_16x16x32_bf16(false, A, false, B3, (short)0, acc[3], false, false);
        A = An; B0 = Bn0; B1 = Bn1; B2 = Bn2; B3 = Bn3;
    }
#undef LD_A
#undef LD_B

    // epilogue: D VGPR r -> M = 8*hi + r, N = lo; add fused bias
#pragma unroll
    for (int j = 0; j < 4; j++) {
        int n = n0 + j * 16 + lo;
        float bs = b_ih[n] + b_hh[n];
#pragma unroll
        for (int r = 0; r < 8; r++) {
            int m = m0 + hi * 8 + r;
            Gx[(size_t)m * GATES + n] = acc[j][r] + bs;
        }
    }
}

// ---------------- Phase 2: recurrence ----------------
// 8 WGs x 256 threads. WG g owns h[j0..j0+64) -> 256 rows of W_hh as 4
// contiguous 64-row blocks (i/f/g/o quadrants), TDM-loaded into LDS with
// 16B/row padding (row stride 1040B -> consecutive rows land 4 banks apart,
// so per-lane A-fragment b128 reads are conflict free).
__global__ void __launch_bounds__(256) lstm_recurrent(
    const float* __restrict__ Gx, const __bf16* __restrict__ Whh,
    const float* __restrict__ c0,
    __bf16* __restrict__ hbuf,      // [2][512] bf16, double-buffered by t&1
    int* __restrict__ cnt,          // [SEQ_LEN+1] arrival counters (pre-zeroed)
    float* __restrict__ out)        // [512]
{
    extern __shared__ char smem[];
    char*   Wlds = smem;                                        // 4*66560 = 260KB
    float*  gbuf = (float*)(smem + 4 * QBLK_BYTES);             // 256 pre-act gates
    __bf16* hlds = (__bf16*)(smem + 4 * QBLK_BYTES + 1024);     // 512 bf16 h_t
    float*  clds = (float*)(smem + 4 * QBLK_BYTES + 2048);      // 64 cell states

    const int tid  = threadIdx.x;
    const int wave = tid >> 5, lane = tid & 31;
    const int lo = lane & 15, hi = lane >> 4;
    const int j0 = blockIdx.x * JPER;

    // -- prologue: TDM the 4 quadrant blocks (64 rows x 512 bf16) into LDS --
    if (wave == 0) {
        unsigned lds_base = (unsigned)(unsigned long long)(uintptr_t)(void*)Wlds;
#pragma unroll
        for (int q = 0; q < 4; q++) {
            unsigned long long ga =
                (unsigned long long)(uintptr_t)(const void*)(Whh + (size_t)(q * HID_DIM + j0) * HID_DIM);
            v4u g0;
            g0[0] = 1u;                                          // count=1, gather off
            g0[1] = lds_base + (unsigned)(q * QBLK_BYTES);       // lds_addr
            g0[2] = (unsigned)(ga & 0xFFFFFFFFu);                // global_addr lo
            g0[3] = (unsigned)((ga >> 32) & 0x1FFFFFFu) | (2u << 30);  // addr hi | type=2
            v8i g1;
            g1[0] = (int)((1u << 16)      // data_size = 2 bytes
                        | (1u << 20)      // pad_enable
                        | (7u << 22)      // pad_interval = 256 DWORDs (one row)
                        | (3u << 25));    // pad_amount   = 4 DWORDs (16B)
            g1[1] = (int)((unsigned)HID_DIM << 16);   // tensor_dim0 = 512 (lo16 @ [31:16])
            g1[2] = (int)((unsigned)64 << 16);        // tensor_dim0 hi=0 | tensor_dim1 = 64
            g1[3] = (int)((unsigned)HID_DIM << 16);   // tensor_dim1 hi=0 | tile_dim0 = 512
            g1[4] = (int)64u;                         // tile_dim1 = 64 | tile_dim2 = 0
            g1[5] = (int)(unsigned)HID_DIM;           // tensor_dim0_stride = 512
            g1[6] = 0; g1[7] = 0;                     // stride hi / tensor_dim1_stride
            v4i gz4; gz4[0] = 0; gz4[1] = 0; gz4[2] = 0; gz4[3] = 0;
            v8i gz8;
#pragma unroll
            for (int i = 0; i < 8; i++) gz8[i] = 0;
            // 6-arg form (clang-23 / therock-10.0 headers)
            __builtin_amdgcn_tensor_load_to_lds(g0, g1, gz4, gz4, gz8, 0);
        }
        __builtin_amdgcn_s_wait_tensorcnt(0);
    }
    if (tid < JPER) clds[tid] = c0[j0 + tid];
    __syncthreads();

    // per-lane A-fragment base addresses for this wave's two M-tiles
    // element (local row lr, k): LDS byte = (lr>>6)*QBLK + (lr&63)*1040 + 2k
    // fragment chunk k-offsets: {kt*32 + hi*8 .. +8} and {+16 .. +24}
    auto fragBase = [&](int m) -> const char* {
        int q = m >> 2, rin = (m & 3) * 16 + lo;
        return Wlds + q * QBLK_BYTES + rin * ROW_BYTES + hi * 16;
    };
    const char* a0base = fragBase(2 * wave);
    const char* a1base = fragBase(2 * wave + 1);

#define LD_A0(kt) make_v16(*(const v8bf*)(a0base + (kt) * 64), *(const v8bf*)(a0base + (kt) * 64 + 32))
#define LD_A1(kt) make_v16(*(const v8bf*)(a1base + (kt) * 64), *(const v8bf*)(a1base + (kt) * 64 + 32))
#define LD_BH(kt) make_v16(*(const v8bf*)(hlds + (kt) * 32 + hi * 8), \
                           *(const v8bf*)(hlds + (kt) * 32 + hi * 8 + 16))

    for (int t = 0; t < SEQ_LEN; ++t) {
        // wait until all 8 WGs published h_t (t=0: written by h_init kernel)
        if (t > 0 && tid == 0) {
            while (__hip_atomic_load(&cnt[t], __ATOMIC_RELAXED,
                                     __HIP_MEMORY_SCOPE_AGENT) < RWG) {}
        }
        __syncthreads();
        __threadfence();

        // stage h_t into LDS (512 bf16)
        const ushort* hsrc = (const ushort*)(hbuf + (size_t)(t & 1) * HID_DIM);
        ((ushort*)hlds)[tid]       = hsrc[tid];
        ((ushort*)hlds)[tid + 256] = hsrc[tid + 256];
        __syncthreads();

        if (lane == 0)   // prefetch next step's pre-gates
            __builtin_prefetch(Gx + (size_t)(t + 1) * GATES + wave * 256, 0, 1);

        // matvec via WMMA: h broadcast into every B column -> D replicated over N
        v8f acc0, acc1;
#pragma unroll
        for (int r = 0; r < 8; r++) { acc0[r] = 0.0f; acc1[r] = 0.0f; }

        v16bf B = LD_BH(0), A0 = LD_A0(0), A1 = LD_A1(0);
#pragma unroll
        for (int kt = 0; kt < 16; ++kt) {
            v16bf Bn = B, A0n = A0, A1n = A1;
            if (kt < 15) { Bn = LD_BH(kt + 1); A0n = LD_A0(kt + 1); A1n = LD_A1(kt + 1); }
            acc0 = __builtin_amdgcn_wmma_f32_16x16x32_bf16(
                false, A0, false, B, (short)0, acc0, false, false);
            acc1 = __builtin_amdgcn_wmma_f32_16x16x32_bf16(
                false, A1, false, B, (short)0, acc1, false, false);
            B = Bn; A0 = A0n; A1 = A1n;
        }

        // lanes 0/16 publish this wave's 32 gate rows (+ Gx[t]) to LDS
        if (lo == 0) {
            const float* gx = Gx + (size_t)t * GATES;
#pragma unroll
            for (int r = 0; r < 8; r++) {
                int lr0 = wave * 32 + hi * 8 + r;
                int lr1 = lr0 + 16;
                gbuf[lr0] = acc0[r] + gx[(lr0 >> 6) * HID_DIM + j0 + (lr0 & 63)];
                gbuf[lr1] = acc1[r] + gx[(lr1 >> 6) * HID_DIM + j0 + (lr1 & 63)];
            }
        }
        __syncthreads();

        // elementwise LSTM cell for the 64 owned h indices
        if (tid < JPER) {
            float iv = sigmf(gbuf[tid]);
            float fv = sigmf(gbuf[64 + tid]);
            float gv = tanhf(gbuf[128 + tid]);
            float ov = sigmf(gbuf[192 + tid]);
            float c  = fv * clds[tid] + iv * gv;
            float h  = ov * tanhf(c);
            clds[tid] = c;
            hbuf[(size_t)((t + 1) & 1) * HID_DIM + j0 + tid] = (__bf16)h;
            if (t == SEQ_LEN - 1) out[j0 + tid] = h;
        }
        __threadfence();        // each writer makes its h stores agent-visible
        __syncthreads();        // all fences done before the arrival signal
        if (tid == 0)
            __hip_atomic_fetch_add(&cnt[t + 1], 1, __ATOMIC_RELEASE,
                                   __HIP_MEMORY_SCOPE_AGENT);
    }
#undef LD_A0
#undef LD_A1
#undef LD_BH
}

// ---------------- launch ----------------
extern "C" void kernel_launch(void* const* d_in, const int* in_sizes, int n_in,
                              void* d_out, int out_size, void* d_ws, size_t ws_size,
                              hipStream_t stream) {
    const int*   x    = (const int*)d_in[0];
    const float* emb  = (const float*)d_in[1];
    const float* W_ih = (const float*)d_in[2];
    const float* W_hh = (const float*)d_in[3];
    const float* b_ih = (const float*)d_in[4];
    const float* b_hh = (const float*)d_in[5];
    const float* h0   = (const float*)d_in[6];
    const float* c0   = (const float*)d_in[7];
    float* out = (float*)d_out;

    // workspace layout (power-of-two aligned); total ~74.5 MB
    char* ws = (char*)d_ws;
    float*  Gx     = (float*)ws;                                    // 64 MB
    __bf16* Wih_bf = (__bf16*)(ws + 67108864);                      // 1 MB
    __bf16* Whh_bf = (__bf16*)(ws + 67108864 + 1048576);            // 2 MB
    __bf16* Ebf    = (__bf16*)(ws + 67108864 + 1048576 + 2097152);  // 4 MB
    __bf16* hbuf   = (__bf16*)(ws + 67108864 + 1048576 + 2097152 + 4194304); // 2 KB
    int*    cnt    = (int*)  (ws + 67108864 + 1048576 + 2097152 + 4194304 + 4096);

    (void)hipMemsetAsync(cnt, 0, (SEQ_LEN + 1) * sizeof(int), stream);
    cvt_bf16<<<(GATES * EMB_DIM + 255) / 256, 256, 0, stream>>>(W_ih, Wih_bf, GATES * EMB_DIM);
    cvt_bf16<<<(GATES * HID_DIM + 255) / 256, 256, 0, stream>>>(W_hh, Whh_bf, GATES * HID_DIM);
    embed_gather<<<SEQ_LEN, 64, 0, stream>>>(x, emb, Ebf);
    h_init<<<1, HID_DIM, 0, stream>>>(h0, hbuf);

    dim3 gg(SEQ_LEN / 128, GATES / 64);
    gemm_pregates<<<gg, 256, 0, stream>>>(Ebf, Wih_bf, b_ih, b_hh, Gx);

    size_t lds = 4 * QBLK_BYTES + 1024 + 1024 + 256;  // padded W + gbuf + hlds + clds
    lstm_recurrent<<<RWG, 256, lds, stream>>>(Gx, Whh_bf, c0, hbuf, cnt, out);
}
